// AssociativeLayerWrapper_41644002902144
// MI455X (gfx1250) — matrix operand order repaired and checked by reading, not compiled, not run on hardware
//
#include <hip/hip_runtime.h>
#include <hip/hip_bf16.h>
#include <math.h>

#define D_DIM 1024
#define F_DIM 192
#define S_LEN 4096
#define B_SZ 4
#define M_TOK 64
#define R_TOT (B_SZ * S_LEN)   // 16384
#define RM_TOT (B_SZ * M_TOK)  // 256
#define EPS_F 1e-5f

typedef __attribute__((ext_vector_type(16))) __bf16 v16bf;
typedef __attribute__((ext_vector_type(8)))  float  v8f;
typedef __attribute__((ext_vector_type(2)))  unsigned int u2;
typedef __attribute__((ext_vector_type(4)))  unsigned int u4;
typedef __attribute__((ext_vector_type(4)))  float  f4;
typedef __attribute__((ext_vector_type(4)))  unsigned int v4u;
typedef __attribute__((ext_vector_type(4)))  int    v4i;
typedef __attribute__((ext_vector_type(8)))  int    v8i;

#if defined(__has_builtin)
# if __has_builtin(__builtin_amdgcn_tensor_load_to_lds) && __has_builtin(__builtin_amdgcn_s_wait_tensorcnt)
#  define HAVE_TDM 1
# endif
#endif

union Frag {
  v16bf v;
  unsigned int u[8];
};

__device__ __forceinline__ unsigned short f2bf(float f) {
  union { float f; unsigned int u; } x; x.f = f;
  unsigned int r = x.u + 0x7fffu + ((x.u >> 16) & 1u);  // RNE
  return (unsigned short)(r >> 16);
}
__device__ __forceinline__ float bf2f(unsigned short h) {
  union { float f; unsigned int u; } x; x.u = ((unsigned int)h) << 16;
  return x.f;
}

// ---- CDNA5 WMMA fragment loaders (wave32) --------------------------------
// A: 16x32 bf16, M x K.  Lanes 0-15: M=lane, K pairs {0,1..6,7 | 16..23}
//                        Lanes 16-31: M=lane-16, K pairs shifted by +8.
__device__ __forceinline__ void load_a(const unsigned short* lds, int lda, int kb, Frag& a) {
  int lane = threadIdx.x & 31;
  int m = lane & 15;
  int kh = (lane < 16) ? 0 : 8;
  const unsigned short* row = lds + m * lda;
#pragma unroll
  for (int r = 0; r < 8; ++r) {
    int k = kb + kh + ((r < 4) ? (2 * r) : (16 + 2 * (r - 4)));
    a.u[r] = *(const unsigned int*)(row + k);  // bf16 pair (k, k+1)
  }
}

// B: 32x16 bf16, K x N.  Lanes 0-15: N=lane, K=0..7,16..23; lanes 16-31: N=lane-16, K=8..15,24..31.
// LDS tile stored pair-interleaved: (k, n) at ((k>>1)*BN + n)*2 + (k&1),
// so each lane reads 8 conflict-free b32 words.
__device__ __forceinline__ void load_b(const unsigned short* lds, int BN, int kb, int n0, Frag& b) {
  int lane = threadIdx.x & 31;
  int n = n0 + (lane & 15);
  int p0 = (kb >> 1) + ((lane < 16) ? 0 : 4);
#pragma unroll
  for (int r = 0; r < 8; ++r) {
    b.u[r] = *(const unsigned int*)(lds + ((p0 + r + ((r < 4) ? 0 : 4)) * BN + n) * 2);
  }
}

__device__ __forceinline__ v8f wmma_bf(const Frag& a, const Frag& b, v8f c) {
  return __builtin_amdgcn_wmma_f32_16x16x32_bf16(
      /*neg_a=*/false, a.v, /*neg_b=*/false, b.v,
      /*c_mod=*/(short)0, c, /*reuse_a=*/false, /*reuse_b=*/false);
}

// Stage a KT x BN bf16 B-tile (row-major in global, stride ldg) into the
// pair-interleaved LDS layout: per item, 2x b64 global loads + 1x b128 LDS store.
__device__ __forceinline__ void stage_b(const unsigned short* __restrict__ gB, size_t ldg,
                                        int kb, int c0, unsigned short* lds, int BN, int KT,
                                        int tid, int nthr) {
  int cg = BN >> 2;            // column groups of 4
  int total = (KT >> 1) * cg;  // k-pairs * col-groups
  for (int e = tid; e < total; e += nthr) {
    int p = e / cg;
    int n4 = (e - p * cg) << 2;
    const unsigned short* gp = gB + (size_t)(kb + 2 * p) * ldg + (size_t)c0 + n4;
    u2 ra = *(const u2*)gp;          // row k,   cols n4..n4+3
    u2 rc = *(const u2*)(gp + ldg);  // row k+1, cols n4..n4+3
    u4 w;
    w.x = (ra.x & 0xffffu) | (rc.x << 16);
    w.y = (ra.x >> 16)     | (rc.x & 0xffff0000u);
    w.z = (ra.y & 0xffffu) | (rc.y << 16);
    w.w = (ra.y >> 16)     | (rc.y & 0xffff0000u);
    *(u4*)(lds + ((p * BN + n4) << 1)) = w;
  }
}

// ---- K0: convert / transpose weights to bf16 -----------------------------
__global__ __launch_bounds__(256) void k_convert(
    const float* __restrict__ Wmq, const float* __restrict__ Wmk,
    const float* __restrict__ Wmem, const float* __restrict__ Wmv,
    unsigned short* __restrict__ WmqT, unsigned short* __restrict__ WmkT,
    unsigned short* __restrict__ Wmem_bf, unsigned short* __restrict__ WmvT) {
  int tid = blockIdx.x * blockDim.x + threadIdx.x;
  int nthr = gridDim.x * blockDim.x;
  for (int e = tid; e < D_DIM * 32; e += nthr) {       // (D x 32), k-major
    int k = e >> 5, n = e & 31;
    WmqT[e] = f2bf(Wmq[n * D_DIM + k]);
    WmkT[e] = f2bf(Wmk[n * D_DIM + k]);
  }
  for (int e = tid; e < F_DIM * D_DIM; e += nthr)      // (F x D), row-major copy
    Wmem_bf[e] = f2bf(Wmem[e]);
  for (int e = tid; e < D_DIM * D_DIM; e += nthr) {    // W_mv transposed
    int k = e >> 10, n = e & 1023;
    WmvT[e] = f2bf(Wmv[(size_t)n * D_DIM + k]);
  }
}

// ---- K1 / K3a: q = src @ Wᵀ (WMMA), dpfp features + denom (+gate) --------
__global__ __launch_bounds__(256) void k_feat(
    const float* __restrict__ src, const unsigned short* __restrict__ WT,
    const float* __restrict__ z, unsigned short* __restrict__ feat_out,
    float* __restrict__ denom_out, int mode,
    const float* __restrict__ Wmb, const float* __restrict__ Wmb_b,
    float* __restrict__ mb_out) {
  __shared__ unsigned short ldsB[32 * 32];
  __shared__ unsigned short ldsA[128 * 32];
  __shared__ float ldsQ[8 * 16 * 32];
  __shared__ float ldsZ[F_DIM];

  int tid = threadIdx.x, wave = tid >> 5, lane = tid & 31;
  int rb = blockIdx.x * 128;
  if (tid < F_DIM) ldsZ[tid] = z[tid];

  v8f acc0 = {}, acc1 = {};
  for (int kb = 0; kb < D_DIM; kb += 32) {
    stage_b(WT, 32, kb, 0, ldsB, 32, 32, tid, 256);
    // A: 128 rows x 32 cols fp32 -> bf16, vectorized float4 -> b64
    for (int e = tid; e < 128 * 8; e += 256) {
      int r = e >> 3, g4 = (e & 7) << 2;
      int lr = rb + r;
      size_t sr = mode ? ((size_t)(lr >> 6) * S_LEN + (S_LEN - M_TOK) + (lr & 63))
                       : (size_t)lr;
      const f4* vp = (const f4*)(src + sr * D_DIM + kb + g4);
      f4 v = mode ? *vp : __builtin_nontemporal_load(vp);
      u2 w;
      w.x = (unsigned)f2bf(v.x) | ((unsigned)f2bf(v.y) << 16);
      w.y = (unsigned)f2bf(v.z) | ((unsigned)f2bf(v.w) << 16);
      *(u2*)(ldsA + (r << 5) + g4) = w;
    }
    __syncthreads();
    Frag a, b0, b1;
    load_a(ldsA + (wave * 16) * 32, 32, 0, a);
    load_b(ldsB, 32, 0, 0, b0);
    load_b(ldsB, 32, 0, 16, b1);
    acc0 = wmma_bf(a, b0, acc0);
    acc1 = wmma_bf(a, b1, acc1);
    __syncthreads();
  }

  // scatter q tiles (16x32 per wave) to LDS
  float* qt = ldsQ + wave * 16 * 32;
#pragma unroll
  for (int v = 0; v < 8; ++v) {
    int m = v + ((lane < 16) ? 0 : 8);
    qt[m * 32 + (lane & 15)] = acc0[v];
    qt[m * 32 + 16 + (lane & 15)] = acc1[v];
  }
  __syncthreads();

  // dpfp: x = [relu(q), relu(-q)]; feat[(j-1)*64+i] = x[i]*x[(i-j+64)%64], j=1..3
  int row = lane >> 1;
  int h = lane & 1;
  int lrow = wave * 16 + row;
  size_t orow = (size_t)(rb + lrow);
  const float* q = ldsQ + wave * 16 * 32 + row * 32;
  float x[64];
#pragma unroll
  for (int i = 0; i < 32; ++i) {
    float v = q[i];
    x[i] = v > 0.f ? v : 0.f;
    x[32 + i] = v < 0.f ? -v : 0.f;
  }
  float dsum = 0.f;
  unsigned short* fo = feat_out + orow * F_DIM;
  for (int fp = h * 48; fp < h * 48 + 48; ++fp) {
    int f0i = fp * 2;
    int j0 = (f0i >> 6) + 1, i0 = f0i & 63, i0b = i0 - j0; if (i0b < 0) i0b += 64;
    int f1i = f0i + 1;
    int j1 = (f1i >> 6) + 1, i1 = f1i & 63, i1b = i1 - j1; if (i1b < 0) i1b += 64;
    float m0 = x[i0] * x[i0b];
    float m1 = x[i1] * x[i1b];
    dsum += m0 * ldsZ[f0i] + m1 * ldsZ[f1i];
    *(unsigned int*)(fo + f0i) = (unsigned)f2bf(m0) | ((unsigned)f2bf(m1) << 16);
  }
  dsum += __shfl_xor(dsum, 1, 32);
  if (h == 0) denom_out[orow] = dsum;

  // deterministic sigmoid gate (mode 1): full-row dot, one thread per row
  if (mode && tid < 128) {
    int lr = rb + tid;
    size_t sr = (size_t)(lr >> 6) * S_LEN + (S_LEN - M_TOK) + (lr & 63);
    float dot = 0.f;
    for (int k = 0; k < D_DIM; ++k) dot += src[sr * D_DIM + k] * Wmb[k];
    mb_out[lr] = 1.f / (1.f + __expf(-(dot + Wmb_b[0])));
  }
}

// ---- K2: num = mq @ W_mem, out = num/(denom+eps) + hs --------------------
__global__ __launch_bounds__(256) void k_gemm_out(
    const unsigned short* __restrict__ mqbf, const unsigned short* __restrict__ Wmem_bf,
    const float* __restrict__ denom, const float* __restrict__ hs,
    float* __restrict__ out) {
  __shared__ unsigned short ldsA[64 * F_DIM];   // 24 KB, full-K A tile
  __shared__ unsigned short ldsB[32 * 128];     // 8 KB, per-chunk swizzled B
  int tid = threadIdx.x, wave = tid >> 5, lane = tid & 31;
  int r0 = blockIdx.x * 64;
  int c0 = blockIdx.y * 128;
  int wr = wave >> 1, wc = wave & 1;

#ifdef HAVE_TDM
  // A-tile: 64 contiguous rows x 192 bf16 = one flat 24 KB segment.
  // 1-D Tensor Data Mover descriptor: data_size=2B, tile_dim0=tensor_dim0=12288.
  if (wave == 0) {
    const unsigned NE = 64 * F_DIM;  // 12288 elements
    unsigned ldsa = (unsigned)(unsigned long long)(ldsA);
    unsigned long long ga =
        (unsigned long long)(const void*)(mqbf + (size_t)r0 * F_DIM);
    v4u g0 = { 1u,                                   // count=1 (valid user D#)
               ldsa,                                 // lds_addr
               (unsigned)(ga & 0xffffffffu),         // global_addr[31:0]
               (unsigned)((ga >> 32) & 0x1ffffffu)   // global_addr[56:32]
                 | (2u << 30) };                     // type=2 ("image")
    v8i g1 = { (int)(1u << 16),            // data_size=2B; no multicast/pad
               (int)((NE & 0xffffu) << 16),// tensor_dim0[15:0]
               (int)(NE >> 16),            // tensor_dim0[31:16]; tensor_dim1=0
               (int)((NE & 0xffffu) << 16),// tile_dim0
               0,                          // tile_dim1=0, tile_dim2=0
               (int)NE,                    // tensor_dim0_stride lo
               0, 0 };
    v4i gz = { 0, 0, 0, 0 };
#if __clang_major__ >= 23
    v8i gz8 = { 0, 0, 0, 0, 0, 0, 0, 0 };
    __builtin_amdgcn_tensor_load_to_lds(g0, g1, gz, gz, gz8, 0);
#else
    __builtin_amdgcn_tensor_load_to_lds(g0, g1, gz, gz, 0);
#endif
    __builtin_amdgcn_s_wait_tensorcnt(0);
  }
#else
  {
    const u4* sA = (const u4*)(mqbf + (size_t)r0 * F_DIM);
    u4* dA = (u4*)ldsA;
    for (int e = tid; e < 64 * F_DIM / 8; e += 256) dA[e] = sA[e];
  }
#endif
  // prefetch the residual stream for the epilogue (global_prefetch_b8)
  {
    int pr = r0 + wr * 16 + (lane & 15);
    const float* p = hs + (size_t)pr * D_DIM + c0 + wc * 64 + ((lane < 16) ? 0 : 32);
    __builtin_prefetch(p, 0, 0);
  }

  v8f acc[4] = {};
  for (int kb = 0; kb < F_DIM; kb += 32) {
    __syncthreads();
    stage_b(Wmem_bf, D_DIM, kb, c0, ldsB, 128, 32, tid, 256);
    __syncthreads();
    Frag a;
    load_a(ldsA + (wr * 16) * F_DIM, F_DIM, kb, a);
#pragma unroll
    for (int t = 0; t < 4; ++t) {
      Frag b;
      load_b(ldsB, 128, 0, wc * 64 + t * 16, b);
      acc[t] = wmma_bf(a, b, acc[t]);
    }
  }

#pragma unroll
  for (int t = 0; t < 4; ++t)
#pragma unroll
    for (int v = 0; v < 8; ++v) {
      int gr = r0 + wr * 16 + v + ((lane < 16) ? 0 : 8);
      int gc = c0 + wc * 64 + t * 16 + (lane & 15);
      size_t idx = (size_t)gr * D_DIM + gc;
      float hv = __builtin_nontemporal_load(hs + idx);
      float val = acc[t][v] / (denom[gr] + EPS_F) + hv;
      __builtin_nontemporal_store(val, out + idx);
    }
}

// ---- K3b: mv = mem @ W_mvᵀ - (mk @ W_mem)/(denom2+eps); store bf16(mv*mb) -
__global__ __launch_bounds__(256) void k_mv(
    const float* __restrict__ out, const unsigned short* __restrict__ WmvT_bf,
    const unsigned short* __restrict__ mkbf, const unsigned short* __restrict__ Wmem_bf,
    const float* __restrict__ denom2, const float* __restrict__ mb,
    unsigned short* __restrict__ mvmb_bf) {
  __shared__ unsigned short ldsA[64 * 32];   // 4 KB
  __shared__ unsigned short ldsB[32 * 128];  // 8 KB
  int tid = threadIdx.x, wave = tid >> 5, lane = tid & 31;
  int r0 = blockIdx.x * 64;
  int c0 = blockIdx.y * 128;
  int wr = wave >> 1, wc = wave & 1;

  v8f acc1[4] = {}, acc2[4] = {};
  // GEMM1: new_mv over K = 1024
  for (int kb = 0; kb < D_DIM; kb += 32) {
    __syncthreads();
    for (int e = tid; e < 64 * 8; e += 256) {
      int r = e >> 3, g4 = (e & 7) << 2;
      int rm = r0 + r;
      size_t sr = (size_t)(rm >> 6) * S_LEN + (S_LEN - M_TOK) + (rm & 63);
      f4 v = *(const f4*)(out + sr * D_DIM + kb + g4);
      u2 w;
      w.x = (unsigned)f2bf(v.x) | ((unsigned)f2bf(v.y) << 16);
      w.y = (unsigned)f2bf(v.z) | ((unsigned)f2bf(v.w) << 16);
      *(u2*)(ldsA + (r << 5) + g4) = w;
    }
    stage_b(WmvT_bf, D_DIM, kb, c0, ldsB, 128, 32, tid, 256);
    __syncthreads();
    Frag a;
    load_a(ldsA + (wr * 16) * 32, 32, 0, a);
#pragma unroll
    for (int t = 0; t < 4; ++t) {
      Frag b;
      load_b(ldsB, 128, 0, wc * 64 + t * 16, b);
      acc1[t] = wmma_bf(a, b, acc1[t]);
    }
  }
  // GEMM2: prev_mv over K = 192
  for (int kb = 0; kb < F_DIM; kb += 32) {
    __syncthreads();
    for (int e = tid; e < 64 * 16; e += 256) {
      int r = e >> 4, c = (e & 15) << 1;
      *(unsigned int*)(ldsA + (r << 5) + c) =
          *(const unsigned int*)(mkbf + (size_t)(r0 + r) * F_DIM + kb + c);
    }
    stage_b(Wmem_bf, D_DIM, kb, c0, ldsB, 128, 32, tid, 256);
    __syncthreads();
    Frag a;
    load_a(ldsA + (wr * 16) * 32, 32, 0, a);
#pragma unroll
    for (int t = 0; t < 4; ++t) {
      Frag b;
      load_b(ldsB, 128, 0, wc * 64 + t * 16, b);
      acc2[t] = wmma_bf(a, b, acc2[t]);
    }
  }

#pragma unroll
  for (int t = 0; t < 4; ++t)
#pragma unroll
    for (int v = 0; v < 8; ++v) {
      int rm = r0 + wr * 16 + v + ((lane < 16) ? 0 : 8);
      int gc = c0 + wc * 64 + t * 16 + (lane & 15);
      float mv = acc1[t][v] - acc2[t][v] / (denom2[rm] + EPS_F);
      mvmb_bf[(size_t)rm * D_DIM + gc] = f2bf(mv * mb[rm]);
    }
}

// ---- K3c: W_mem_new[b] = W_mem + mkᵀ @ (mv*mb), K = 64 -------------------
__global__ __launch_bounds__(256) void k_assoc(
    const unsigned short* __restrict__ mkbf, const unsigned short* __restrict__ mvmb_bf,
    const float* __restrict__ Wmem, float* __restrict__ Wmem_new) {
  __shared__ unsigned short ldsA[64 * 66];    // A = mkᵀ (f x m), padded stride
  __shared__ unsigned short ldsB[64 * 128];   // B = mv*mb (m x n), swizzled
  int tid = threadIdx.x, wave = tid >> 5, lane = tid & 31;
  int f0 = blockIdx.x * 64;
  int c0 = blockIdx.y * 128;
  int b  = blockIdx.z;
  int wr = wave >> 1, wc = wave & 1;

  for (int e = tid; e < 64 * 64; e += 256) {   // transpose mk during LDS store
    int m = e >> 6, f = e & 63;
    ldsA[f * 66 + m] = mkbf[(size_t)(b * M_TOK + m) * F_DIM + f0 + f];
  }
  stage_b(mvmb_bf + (size_t)b * M_TOK * D_DIM, D_DIM, 0, c0, ldsB, 128, 64, tid, 256);
  __syncthreads();

  v8f acc[4] = {};
#pragma unroll
  for (int kb = 0; kb < 64; kb += 32) {
    Frag a;
    load_a(ldsA + (wr * 16) * 66, 66, kb, a);
#pragma unroll
    for (int t = 0; t < 4; ++t) {
      Frag bb;
      load_b(ldsB, 128, kb, wc * 64 + t * 16, bb);
      acc[t] = wmma_bf(a, bb, acc[t]);
    }
  }

#pragma unroll
  for (int t = 0; t < 4; ++t)
#pragma unroll
    for (int v = 0; v < 8; ++v) {
      int f = f0 + wr * 16 + v + ((lane < 16) ? 0 : 8);
      int gc = c0 + wc * 64 + t * 16 + (lane & 15);
      Wmem_new[((size_t)b * F_DIM + f) * D_DIM + gc] =
          acc[t][v] + Wmem[(size_t)f * D_DIM + gc];
    }
}

// ---- K3d: z_new = z + sum_m mk ------------------------------------------
__global__ __launch_bounds__(256) void k_znew(
    const unsigned short* __restrict__ mkbf, const float* __restrict__ z,
    float* __restrict__ znew) {
  int idx = blockIdx.x * blockDim.x + threadIdx.x;
  if (idx >= B_SZ * F_DIM) return;
  int b = idx / F_DIM, f = idx % F_DIM;
  float s = 0.f;
  for (int m = 0; m < M_TOK; ++m) s += bf2f(mkbf[(size_t)(b * M_TOK + m) * F_DIM + f]);
  znew[idx] = z[f] + s;
}

// --------------------------------------------------------------------------
extern "C" void kernel_launch(void* const* d_in, const int* in_sizes, int n_in,
                              void* d_out, int out_size, void* d_ws, size_t ws_size,
                              hipStream_t stream) {
  (void)in_sizes; (void)n_in; (void)out_size; (void)ws_size;
  const float* hs    = (const float*)d_in[0];
  const float* Wmem  = (const float*)d_in[1];
  const float* z     = (const float*)d_in[2];
  const float* Wmq   = (const float*)d_in[3];
  const float* Wmk   = (const float*)d_in[4];
  const float* Wmv   = (const float*)d_in[5];
  const float* Wmb_w = (const float*)d_in[6];
  const float* Wmb_b = (const float*)d_in[7];

  float* out      = (float*)d_out;
  float* Wmem_new = out + (size_t)R_TOT * D_DIM;
  float* z_new    = Wmem_new + (size_t)B_SZ * F_DIM * D_DIM;

  size_t off = 0;
  auto wsalloc = [&](size_t bytes) -> void* {
    void* p = (char*)d_ws + off;
    off += (bytes + 255) & ~(size_t)255;
    return p;
  };
  unsigned short* WmqT_bf = (unsigned short*)wsalloc((size_t)D_DIM * 32 * 2);
  unsigned short* WmkT_bf = (unsigned short*)wsalloc((size_t)D_DIM * 32 * 2);
  unsigned short* Wmem_bf = (unsigned short*)wsalloc((size_t)F_DIM * D_DIM * 2);
  unsigned short* WmvT_bf = (unsigned short*)wsalloc((size_t)D_DIM * D_DIM * 2);
  unsigned short* mq_bf   = (unsigned short*)wsalloc((size_t)R_TOT * F_DIM * 2);
  float*          denom   = (float*)wsalloc((size_t)R_TOT * 4);
  unsigned short* mk_bf   = (unsigned short*)wsalloc((size_t)RM_TOT * F_DIM * 2);
  float*          denom2  = (float*)wsalloc((size_t)RM_TOT * 4);
  float*          mb      = (float*)wsalloc((size_t)RM_TOT * 4);
  unsigned short* mvmb_bf = (unsigned short*)wsalloc((size_t)RM_TOT * D_DIM * 2);

  // K0: weights -> bf16 (transposed so every GEMM reads B row-major in K)
  k_convert<<<256, 256, 0, stream>>>(Wmq, Wmk, Wmem, Wmv,
                                     WmqT_bf, WmkT_bf, Wmem_bf, WmvT_bf);
  // K1: mq features + denom (16384 rows)
  k_feat<<<R_TOT / 128, 256, 0, stream>>>(hs, WmqT_bf, z, mq_bf, denom, 0,
                                          Wmb_w, Wmb_b, mb);
  // K2: out = (mq @ W_mem)/(denom+eps) + hs
  k_gemm_out<<<dim3(R_TOT / 64, D_DIM / 128), 256, 0, stream>>>(
      mq_bf, Wmem_bf, denom, hs, out);
  // K3a: mk features + denom2 + sigmoid gate (256 mem rows)
  k_feat<<<RM_TOT / 128, 256, 0, stream>>>(out, WmkT_bf, z, mk_bf, denom2, 1,
                                           Wmb_w, Wmb_b, mb);
  // K3b: mv = mem @ W_mvᵀ - (mk @ W_mem)/(denom2+eps); store bf16(mv*mb)
  k_mv<<<dim3(RM_TOT / 64, D_DIM / 128), 256, 0, stream>>>(
      out, WmvT_bf, mk_bf, Wmem_bf, denom2, mb, mvmb_bf);
  // K3c: W_mem_new = W_mem + mkᵀ @ (mv*mb)
  k_assoc<<<dim3(F_DIM / 64, D_DIM / 128, B_SZ), 256, 0, stream>>>(
      mk_bf, mvmb_bf, Wmem, Wmem_new);
  // K3d: z_new = z + mk.sum(m)
  k_znew<<<(B_SZ * F_DIM + 255) / 256, 256, 0, stream>>>(mk_bf, z, z_new);
}
